// PairSelfAttention_45655502356820
// MI455X (gfx1250) — matrix operand order repaired and checked by reading, not compiled, not run on hardware
//
#include <hip/hip_runtime.h>

typedef __attribute__((ext_vector_type(16))) _Float16 v16h;
typedef __attribute__((ext_vector_type(8)))  _Float16 v8h;
typedef __attribute__((ext_vector_type(8)))  float    v8f;

#define SEQ  1024
#define HDIM 64
#define PDIM 128   // paired head dim = 2*HDIM
#define KT   64    // keys per tile
#define QT   128   // q rows per block
#define NWAVE 8

// ---------------------------------------------------------------------------
// WMMA helpers (gfx1250: D = A(16x32 f16) * B(32x16 f16) + C(16x16 f32))
// ---------------------------------------------------------------------------
__device__ __forceinline__ v8f wmma_f16(v16h a, v16h b, v8f c) {
  return __builtin_amdgcn_wmma_f32_16x16x32_f16(false, a, false, b, (short)0, c,
                                                false, false);
}

// A-matrix fragment: lane L (half = L>=16): row fixed, halves[0..7] = K d0..d0+7,
// halves[8..15] = K d0+16..d0+23, where d0 = chunk + half*8   (ISA 7.12.2)
__device__ __forceinline__ v16h frag_a(const _Float16* p) {
  union { v16h v; v8h h[2]; } u;
  u.h[0] = *(const v8h*)(p);
  u.h[1] = *(const v8h*)(p + 16);
  return u.v;
}
// B-matrix fragment: lane L: column N = L&15 fixed, halves[0..7] = K d0..d0+7,
// halves[8..15] = K d0+16..d0+23, where d0 = chunk + half*8   (matches frag_a)
__device__ __forceinline__ v16h frag_b(const _Float16* p) {
  union { v16h v; v8h h[2]; } u;
  u.h[0] = *(const v8h*)(p);
  u.h[1] = *(const v8h*)(p + 16);
  return u.v;
}

// ---------------------------------------------------------------------------
// Kernel 1: IoU partner selection + pair ordering (matches reference, incl.
// unclamped intersection and "-eye" diagonal, first-max argmax semantics).
// ---------------------------------------------------------------------------
__global__ __launch_bounds__(256)
void pairsel_kernel(const float* __restrict__ centers, int2* __restrict__ pairs) {
  __shared__ float bx1[SEQ], by1[SEQ], bx2[SEQ], by2[SEQ], bar[SEQ], bl1[SEQ];
  const int b = blockIdx.x;
  const float* C = centers + (size_t)b * SEQ * 4;

  for (int s = threadIdx.x; s < SEQ; s += 256) {
    float cx = C[s * 4 + 0], cy = C[s * 4 + 1];
    float hh = C[s * 4 + 2], ww = C[s * 4 + 3];
    float x1 = cx - ww * 0.5f, y1 = cy - hh * 0.5f;
    float x2 = cx + ww * 0.5f, y2 = cy + hh * 0.5f;
    bx1[s] = x1; by1[s] = y1; bx2[s] = x2; by2[s] = y2;
    bar[s] = (x2 - x1) * (y2 - y1);
    bl1[s] = fabsf(x2 - x1) + fabsf(y2 - y1);
  }
  __syncthreads();

  for (int s = threadIdx.x; s < SEQ; s += 256) {
    float x1 = bx1[s], y1 = by1[s], x2 = bx2[s], y2 = by2[s], ar = bar[s];
    float best = -1e30f;
    int bi = 0;
    for (int kk = 0; kk < SEQ; ++kk) {
      float ix1 = fmaxf(x1, bx1[kk]);
      float iy1 = fmaxf(y1, by1[kk]);
      float ix2 = fminf(x2, bx2[kk]);
      float iy2 = fminf(y2, by2[kk]);
      float ia  = (ix2 - ix1) * (iy2 - iy1);      // no clamp (match reference)
      float un  = ar + bar[kk] - ia;
      float iou = ia / un - ((kk == s) ? 1.0f : 0.0f);
      if (iou > best) { best = iou; bi = kk; }
    }
    int p0, p1;
    if (bl1[s] >= bl1[bi]) { p0 = s;  p1 = bi; }
    else                   { p0 = bi; p1 = s;  }
    pairs[(size_t)b * SEQ + s] = make_int2(p0, p1);
  }
}

// ---------------------------------------------------------------------------
// Kernel 2: flash attention over paired Q'/K'/V' (d = 128), WMMA f16->f32.
// Block = (b*H + h, q-tile of 128 rows), 256 threads = 8 waves x 16 rows.
// ---------------------------------------------------------------------------
__global__ __launch_bounds__(256, 1)
void pairattn_kernel(const float* __restrict__ Q, const float* __restrict__ K,
                     const float* __restrict__ V, const int2* __restrict__ pairsAll,
                     float* __restrict__ out) {
  // 48KB aliased LDS: [kbuf 16K | vbuf 16K | pbuf 16K]; Q staging reuses [0,32K)
  __shared__ __align__(16) _Float16 smem[24576];
  __shared__ unsigned char flip[QT];
  _Float16* kbuf   = smem;               // [key][dim]  KT x PDIM
  _Float16* vbuf   = smem + 8192;        // [dim][key]  PDIM x KT (transposed)
  _Float16* pbuf   = smem + 16384;       // per-wave 16 x KT probs
  _Float16* qstage = smem;               // prologue only: QT x PDIM

  const int tid  = threadIdx.x;
  const int lane = tid & 31;
  const int wave = tid >> 5;
  const int lh   = lane >> 4;            // which 16-lane half
  const int l16  = lane & 15;

  const int bh = blockIdx.x;             // b*H + h
  const int b  = bh >> 3;
  const int h  = bh & 7;
  const int qbase = blockIdx.y * QT;

  const float* Qb = Q + (size_t)bh * SEQ * HDIM;
  const float* Kb = K + (size_t)bh * SEQ * HDIM;
  const float* Vb = V + (size_t)bh * SEQ * HDIM;
  const int2* pairs = pairsAll + (size_t)b * SEQ;

  // ---- prologue: gather Q' (f32 -> f16) into LDS, row-major [row][dim] ----
  {
    int row  = tid >> 1;                 // 0..127
    int part = tid & 1;                  // 0: p0 dims 0..63, 1: p1 dims 64..127
    int qg = qbase + row;
    int2 pr = pairs[qg];
    int p = part ? pr.y : pr.x;
    const float* src = Qb + (size_t)p * HDIM;
    _Float16* dst = qstage + row * PDIM + part * HDIM;
#pragma unroll
    for (int d0 = 0; d0 < HDIM; d0 += 4) {
      float4 f = *(const float4*)(src + d0);
      union { _Float16 hh[4]; unsigned long long u64; } pk;
      pk.hh[0] = (_Float16)f.x; pk.hh[1] = (_Float16)f.y;
      pk.hh[2] = (_Float16)f.z; pk.hh[3] = (_Float16)f.w;
      *(unsigned long long*)(dst + d0) = pk.u64;
    }
    if (tid < QT) {
      int qg2 = qbase + tid;
      flip[tid] = (pairs[qg2].x != qg2) ? 1 : 0;
    }
  }
  __syncthreads();

  // ---- load Q A-fragments (16 rows x 128 dims = 4 frags), free the staging --
  v16h qfrag[4];
  {
    const int arow = wave * 16 + l16;
#pragma unroll
    for (int dc = 0; dc < 4; ++dc)
      qfrag[dc] = frag_a(qstage + arow * PDIM + dc * 32 + lh * 8);
  }

  // ---- accumulators & softmax state (row r means actual row r + 8*lh) ----
  v8f oacc[8];
  float mrow[8], lrow[8];
#pragma unroll
  for (int i = 0; i < 8; ++i) { v8f z = {}; oacc[i] = z; }
#pragma unroll
  for (int r = 0; r < 8; ++r) { mrow[r] = -1e30f; lrow[r] = 0.f; }

  _Float16* myp = pbuf + wave * 16 * KT;

  for (int it = 0; it < SEQ / KT; ++it) {
    __syncthreads();                     // previous tile fully consumed
    // ---- cooperative gather-convert of K' (row-major) and V' (transposed) --
    {
      int key = tid >> 2;
      int sub = tid & 3;
      int kg = it * KT + key;
      int2 pr = pairs[kg];
      if (sub < 2) {
        int p = sub ? pr.y : pr.x;
        const float* src = Kb + (size_t)p * HDIM;
        _Float16* dst = kbuf + key * PDIM + sub * HDIM;
#pragma unroll
        for (int d0 = 0; d0 < HDIM; d0 += 4) {
          float4 f = *(const float4*)(src + d0);
          union { _Float16 hh[4]; unsigned long long u64; } pk;
          pk.hh[0] = (_Float16)f.x; pk.hh[1] = (_Float16)f.y;
          pk.hh[2] = (_Float16)f.z; pk.hh[3] = (_Float16)f.w;
          *(unsigned long long*)(dst + d0) = pk.u64;
        }
      } else {
        int p = (sub == 3) ? pr.y : pr.x;
        const float* src = Vb + (size_t)p * HDIM;
        int dbase = (sub == 3) ? HDIM : 0;
#pragma unroll 8
        for (int d = 0; d < HDIM; ++d)
          vbuf[(dbase + d) * KT + key] = (_Float16)src[d];
      }
    }
    __syncthreads();

    // ---- S = Q' K'^T : 4 key-subtiles x 4 d-chunks of WMMA ----
    v8f sacc[4];
#pragma unroll
    for (int j = 0; j < 4; ++j) {
      v8f c = {};
#pragma unroll
      for (int dc = 0; dc < 4; ++dc) {
        v16h bf = frag_b(kbuf + (j * 16 + l16) * PDIM + dc * 32 + lh * 8);
        c = wmma_f16(qfrag[dc], bf, c);
      }
      sacc[j] = c;
    }

    // ---- online softmax over this 64-key tile (per-row, width-16 shuffles) --
    float alpha[8];
#pragma unroll
    for (int r = 0; r < 8; ++r) {
      float t = fmaxf(fmaxf(sacc[0][r], sacc[1][r]),
                      fmaxf(sacc[2][r], sacc[3][r]));
#pragma unroll
      for (int off = 8; off; off >>= 1) t = fmaxf(t, __shfl_xor(t, off, 16));
      float mnew = fmaxf(mrow[r], t);
      alpha[r] = __expf(mrow[r] - mnew);
      mrow[r] = mnew;
      float rs = 0.f;
#pragma unroll
      for (int j = 0; j < 4; ++j) {
        float e = __expf(sacc[j][r] - mnew);
        sacc[j][r] = e;
        rs += e;
      }
#pragma unroll
      for (int off = 8; off; off >>= 1) rs += __shfl_xor(rs, off, 16);
      lrow[r] = lrow[r] * alpha[r] + rs;
    }

    // ---- stage P (f16) to per-wave LDS so it can re-enter A-frag layout ----
#pragma unroll
    for (int j = 0; j < 4; ++j)
#pragma unroll
      for (int r = 0; r < 8; ++r)
        myp[(r + 8 * lh) * KT + j * 16 + l16] = (_Float16)sacc[j][r];

    // rescale running O by alpha
#pragma unroll
    for (int i = 0; i < 8; ++i)
#pragma unroll
      for (int r = 0; r < 8; ++r) oacc[i][r] *= alpha[r];

    // ---- O += P @ V' : 2 key-chunks x 8 output dim-chunks ----
    v16h pfrag[2];
#pragma unroll
    for (int kc = 0; kc < 2; ++kc)
      pfrag[kc] = frag_a(myp + l16 * KT + kc * 32 + lh * 8);
#pragma unroll
    for (int nc = 0; nc < 8; ++nc) {
      v8f c = oacc[nc];
#pragma unroll
      for (int kc = 0; kc < 2; ++kc) {
        v16h bf = frag_b(vbuf + (nc * 16 + l16) * KT + kc * 32 + lh * 8);
        c = wmma_f16(pfrag[kc], bf, c);
      }
      oacc[nc] = c;
    }
  }

  // ---- epilogue: normalize, route heads 0-3 vs 4-7 by flip flag, store ----
  const float post = 0.08838834764831843f;   // 1/sqrt(2*D)
#pragma unroll
  for (int r = 0; r < 8; ++r) {
    int rowm = r + 8 * lh;
    int qg = qbase + wave * 16 + rowm;
    bool f = flip[wave * 16 + rowm] != 0;
    bool wr = (h < 4) ? (!f) : f;
    if (wr) {
      float sc = post / lrow[r];
      size_t obase = ((size_t)(b * SEQ + qg)) * 512 + (size_t)(h & 3) * 128 + l16;
#pragma unroll
      for (int nc = 0; nc < 8; ++nc)
        out[obase + (size_t)nc * 16] = oacc[nc][r] * sc;
    }
  }
}

// ---------------------------------------------------------------------------
extern "C" void kernel_launch(void* const* d_in, const int* in_sizes, int n_in,
                              void* d_out, int out_size, void* d_ws, size_t ws_size,
                              hipStream_t stream) {
  (void)in_sizes; (void)n_in; (void)out_size; (void)ws_size;
  const float* q       = (const float*)d_in[0];
  const float* k       = (const float*)d_in[1];
  const float* v       = (const float*)d_in[2];
  const float* centers = (const float*)d_in[3];
  float* out = (float*)d_out;
  int2* pairs = (int2*)d_ws;               // 16*1024 int2 = 128 KB scratch

  pairsel_kernel<<<16, 256, 0, stream>>>(centers, pairs);
  pairattn_kernel<<<dim3(16 * 8, SEQ / QT), 256, 0, stream>>>(q, k, v, pairs, out);
}
